// AttentionOnly_54391465836544
// MI455X (gfx1250) — matrix operand order repaired and checked by reading, not compiled, not run on hardware
//
#include <hip/hip_runtime.h>
#include <stdint.h>

typedef __attribute__((ext_vector_type(16))) _Float16     v16h;
typedef __attribute__((ext_vector_type(8)))  float        v8f;
typedef __attribute__((ext_vector_type(4)))  float        f4;
typedef __attribute__((ext_vector_type(4)))  unsigned int v4u;
typedef __attribute__((ext_vector_type(8)))  int          v8i;
typedef __attribute__((ext_vector_type(4)))  int          v4i;

namespace {
constexpr int   kHeads        = 32;
constexpr int   kHeadDim      = 128;
constexpr int   kKvLen        = 4096;
constexpr int   kBatch        = 8;
constexpr int   kWaves        = 8;                       // 256 threads / wave32
constexpr int   kTile         = 16;                      // keys per WMMA tile
constexpr int   kKeysPerWave  = kKvLen / kWaves;         // 512
constexpr int   kTilesPerWave = kKeysPerWave / kTile;    // 32
constexpr int   kPitch        = 132;                     // 128 + 4 pad DWORDs (TDM pad)
constexpr float kScale        = 0.088388347648318447f;   // 128^-0.5
}

// Issue one TDM 2D tile load: 16 rows x 128 f32 from global -> LDS, with the
// TDM inserting 4 pad DWORDs after every 128 DWORDs (row pitch 132 DWORDs) so
// later per-row ds_load_b128 reads are bank-conflict free.
__device__ __forceinline__ void tdm_load_k_tile(const float* gptr,
                                                uint32_t lds_byte_addr) {
  const uint64_t ga = (uint64_t)(uintptr_t)gptr;
  v4u g0;
  g0[0] = 1u;                                   // count=1 (valid user descriptor)
  g0[1] = lds_byte_addr;                        // D#.lds_addr (bytes)
  g0[2] = (uint32_t)ga;                         // global_addr[31:0]
  g0[3] = (uint32_t)((ga >> 32) & 0x01FFFFFFu)  // global_addr[56:32]
          | (2u << 30);                         // type=2 ("image")
  v8i g1;
  g1[0] = (2 << 16)     // data_size = 4 bytes
        | (1 << 20)     // pad_enable
        | (6 << 22)     // pad_interval code 6 -> every 128 DWORDs
        | (3 << 25);    // pad_amount  code 3 -> 4 DWORDs
  g1[1] = (int)(128u << 16);   // tensor_dim0 = 128 (bits 79:48)
  g1[2] = (int)(16u << 16);    // tensor_dim1 = 16  (bits 111:80)
  g1[3] = (int)(128u << 16);   // tile_dim0  = 128  (bits 127:112)
  g1[4] = 16;                  // tile_dim1  = 16   (bits 143:128); tile_dim2=0
  g1[5] = 128;                 // tensor_dim0_stride = 128 (bits 207:160)
  g1[6] = 0;                   // stride hi / tensor_dim1_stride = 0 (2D tile)
  g1[7] = 0;
  v4i z4 = {0, 0, 0, 0};       // groups 2/3 unused (tile_dim2..4 = 0)
#if __clang_major__ >= 23
  v8i z8 = {0, 0, 0, 0, 0, 0, 0, 0};
  __builtin_amdgcn_tensor_load_to_lds(g0, g1, z4, z4, z8, 0);
#else
  __builtin_amdgcn_tensor_load_to_lds(g0, g1, z4, z4, 0);
#endif
}

// One workgroup per (batch, head). 8 waves; wave w owns keys [w*512, (w+1)*512).
// Per wave: TDM double-buffers K tiles into LDS (TENSORcnt pipelined), QK^T via
// v_wmma_f32_16x16x32_f16, online softmax, coalesced-V P*V; waves merge via
// log-sum-exp in LDS at the end.
__global__ __launch_bounds__(256) void attn_decode_kernel(
    const float* __restrict__ q, const float* __restrict__ k,
    const float* __restrict__ v, const float* __restrict__ mask,
    float* __restrict__ out) {
  const int lane = threadIdx.x & 31;
  const int wave = threadIdx.x >> 5;
  const int half = lane >> 4;   // which K-subset this lane carries in A/B layouts
  const int m16  = lane & 15;   // A-matrix row (key within tile)

  const int bh = blockIdx.x;                 // b*H + h
  const int b  = bh / kHeads;
  const float* __restrict__ qp = q + (size_t)bh * kHeadDim;
  const float* __restrict__ kp = k + (size_t)bh * kKvLen * kHeadDim;
  const float* __restrict__ vp = v + (size_t)bh * kKvLen * kHeadDim;
  const float* __restrict__ mp = mask + (size_t)b * kKvLen;

  // Per-wave double-buffered K staging (TDM destination), padded rows.
  __shared__ __align__(16) float kstage[kWaves][2][kTile * kPitch];
  __shared__ float sm_s[kWaves];
  __shared__ float sl_s[kWaves];
  __shared__ __align__(16) float so_s[kWaves][kHeadDim];

  // ---- q (pre-scaled) in B-matrix layout: 4 chunks of 32 dims -------------
  // B is 32x16 f16; all 16 columns get the same q chunk so every column of the
  // WMMA result is the score vector. Lanes 0-15 carry K={0..7,16..23} of the chunk,
  // lanes 16-31 carry K={8..15,24..31}.
  v16h bq[4];
#pragma unroll
  for (int c = 0; c < 4; ++c) {
    const float* src = qp + 32 * c + 8 * half;
    f4 q0 = *(const f4*)(src);
    f4 q1 = *(const f4*)(src + 4);
    f4 q2 = *(const f4*)(src + 16);
    f4 q3 = *(const f4*)(src + 20);
#pragma unroll
    for (int i = 0; i < 4; ++i) {
      bq[c][i]      = (_Float16)(q0[i] * kScale);
      bq[c][4 + i]  = (_Float16)(q1[i] * kScale);
      bq[c][8 + i]  = (_Float16)(q2[i] * kScale);
      bq[c][12 + i] = (_Float16)(q3[i] * kScale);
    }
  }

  float m_run = -1e30f;
  float l_run = 0.0f;
  f4 o = {0.0f, 0.0f, 0.0f, 0.0f};   // this lane's 4 output dims: [4*lane, 4*lane+4)

  const int keystart = wave * kKeysPerWave;

  // Prologue: start DMA of tile 0 into buffer 0.
  tdm_load_k_tile(kp + (size_t)keystart * kHeadDim,
                  (uint32_t)(uintptr_t)&kstage[wave][0][0]);

#pragma unroll 1
  for (int t = 0; t < kTilesPerWave; ++t) {
    const int kb = keystart + t * kTile;

    // ---- kick off next tile's K DMA; prefetch next V lines ----------------
    if (t + 1 < kTilesPerWave) {
      tdm_load_k_tile(kp + (size_t)(kb + kTile) * kHeadDim,
                      (uint32_t)(uintptr_t)&kstage[wave][(t + 1) & 1][0]);
      const float* vn = vp + (size_t)(kb + kTile) * kHeadDim;
      __builtin_prefetch(vn + lane * 32, 0, 3);          // lines 0..31 of 8KB tile
      __builtin_prefetch(vn + 1024 + lane * 32, 0, 3);   // lines 32..63
      __builtin_amdgcn_s_wait_tensorcnt(1);  // tile t's DMA (older) is done
    } else {
      __builtin_amdgcn_s_wait_tensorcnt(0);
    }

    // ---- QK^T scores for 16 keys via 4x v_wmma_f32_16x16x32_f16 -----------
    // A (16x32 f16) layout: lane half=0 carries K {0..7, 16..23} of the chunk,
    // half=1 carries {8..15, 24..31}; row = m16. Reads from padded LDS rows:
    // pitch 132 DWORDs -> the 16 lanes of a half hit 16 distinct banks.
    const float* lrow = &kstage[wave][t & 1][m16 * kPitch];
    v8f acc = {};
#pragma unroll
    for (int c = 0; c < 4; ++c) {
      const float* s0 = lrow + 32 * c + 8 * half;        // local K 0..7 / 8..15
      const float* s1 = s0 + 16;                         // local K 16..23 / 24..31
      f4 k0 = *(const f4*)(s0);
      f4 k1 = *(const f4*)(s0 + 4);
      f4 k2 = *(const f4*)(s1);
      f4 k3 = *(const f4*)(s1 + 4);
      v16h a;
#pragma unroll
      for (int i = 0; i < 4; ++i) {
        a[i]      = (_Float16)k0[i];
        a[4 + i]  = (_Float16)k1[i];
        a[8 + i]  = (_Float16)k2[i];
        a[12 + i] = (_Float16)k3[i];
      }
      acc = __builtin_amdgcn_wmma_f32_16x16x32_f16(
          false, a, false, bq[c], (short)0, acc, false, false);
    }

    // C/D layout: VGPR r holds key r on lanes 0-15, key 8+r on lanes 16-31
    // (all 16 columns identical). One SWAPX16 swizzle gives every lane all 16.
    float sc[16];
#pragma unroll
    for (int r = 0; r < 8; ++r) {
      float own = acc[r];
      float oth = __int_as_float(
          __builtin_amdgcn_ds_swizzle(__float_as_int(own), 0x401F));  // SWAPX16
      sc[r]     = half ? oth : own;   // keys 0..7
      sc[8 + r] = half ? own : oth;   // keys 8..15
    }

    // additive mask
    f4 mk0 = *(const f4*)(mp + kb);
    f4 mk1 = *(const f4*)(mp + kb + 4);
    f4 mk2 = *(const f4*)(mp + kb + 8);
    f4 mk3 = *(const f4*)(mp + kb + 12);
#pragma unroll
    for (int i = 0; i < 4; ++i) {
      sc[i] += mk0[i]; sc[4 + i] += mk1[i];
      sc[8 + i] += mk2[i]; sc[12 + i] += mk3[i];
    }

    // ---- online softmax update (uniform across lanes) ---------------------
    float tmax = sc[0];
#pragma unroll
    for (int m = 1; m < 16; ++m) tmax = fmaxf(tmax, sc[m]);
    const float m_new = fmaxf(m_run, tmax);
    const float alpha = __expf(m_run - m_new);
    float p[16];
    float psum = 0.0f;
#pragma unroll
    for (int m = 0; m < 16; ++m) {
      p[m] = __expf(sc[m] - m_new);
      psum += p[m];
    }
    l_run = l_run * alpha + psum;
    m_run = m_new;

    // ---- P·V: lane owns dims [4*lane, 4*lane+4); loads are 512B coalesced -
    o = o * alpha;
    const float* vrow = vp + (size_t)kb * kHeadDim + 4 * lane;
#pragma unroll
    for (int m = 0; m < 16; ++m) {
      f4 vv = *(const f4*)(vrow + m * kHeadDim);
      o += vv * p[m];
    }
  }

  // ---- cross-wave log-sum-exp merge through LDS ----------------------------
  *(f4*)&so_s[wave][4 * lane] = o;
  if (lane == 0) {
    sm_s[wave] = m_run;
    sl_s[wave] = l_run;
  }
  __syncthreads();

  if (threadIdx.x < kHeadDim) {
    const int d = threadIdx.x;
    float M = sm_s[0];
#pragma unroll
    for (int w = 1; w < kWaves; ++w) M = fmaxf(M, sm_s[w]);
    float L = 0.0f, accum = 0.0f;
#pragma unroll
    for (int w = 0; w < kWaves; ++w) {
      const float e = __expf(sm_s[w] - M);
      L += sl_s[w] * e;
      accum += e * so_s[w][d];
    }
    out[(size_t)bh * kHeadDim + d] = accum / L;
  }
}

extern "C" void kernel_launch(void* const* d_in, const int* in_sizes, int n_in,
                              void* d_out, int out_size, void* d_ws, size_t ws_size,
                              hipStream_t stream) {
  (void)in_sizes; (void)n_in; (void)d_ws; (void)ws_size; (void)out_size;
  const float* q    = (const float*)d_in[0];
  const float* k    = (const float*)d_in[1];
  const float* v    = (const float*)d_in[2];
  const float* mask = (const float*)d_in[3];
  float* out        = (float*)d_out;

  dim3 grid(kBatch * kHeads);   // 256 workgroups, one per (b,h)
  dim3 block(256);              // 8 wave32 waves
  attn_decode_kernel<<<grid, block, 0, stream>>>(q, k, v, mask, out);
}
